// FishNet_MaskOutput_28673201668369
// MI455X (gfx1250) — compile-verified, matrix operand order checked by reading, not run on hardware
//
#include <hip/hip_runtime.h>

// ---------------- problem constants ----------------
#define BATCH 200
#define SEQT  500
#define DIN1  300
#define DINP1 320      // x inner dim padded to mult-of-32 (bf16, zero pad)
#define HID1  256
#define HPAD1 256
#define HID2  300
#define HPAD2 320      // h2 padded (pad cols stay zero)
#define KROW  576      // WcatT row length: DINP+HPAD = 576 for both layers

typedef __bf16 bf16;
typedef unsigned int uint32;
typedef __attribute__((ext_vector_type(16))) __bf16  v16bf;
typedef __attribute__((ext_vector_type(8)))  float   v8f;
typedef __attribute__((ext_vector_type(8)))  uint32  v8u;

__device__ __forceinline__ float sigmoidf_(float x) { return 1.f / (1.f + __expf(-x)); }

// WcatT[j][k] row-major [4H, KROW]: row j = [ Wih[j][0..Din) | 0.. | Whh[j][0..H) | 0.. ]
__global__ void prep_wcatT_kernel(const float* __restrict__ Wih,
                                  const float* __restrict__ Whh,
                                  int Din, int DinPad, int H,
                                  bf16* __restrict__ W) {
  int idx = blockIdx.x * blockDim.x + threadIdx.x;
  int total = 4 * H * KROW;
  if (idx >= total) return;
  int j = idx / KROW;
  int k = idx - j * KROW;
  float v = 0.f;
  if (k < Din)                              v = Wih[(size_t)j * Din + k];
  else if (k >= DinPad && k < DinPad + H)   v = Whh[(size_t)j * H + (k - DinPad)];
  W[idx] = (bf16)v;
}

__global__ void prep_bias_kernel(const float* __restrict__ bih,
                                 const float* __restrict__ bhh,
                                 int N4H, float* __restrict__ bsum) {
  int idx = blockIdx.x * blockDim.x + threadIdx.x;
  if (idx < N4H) bsum[idx] = bih[idx] + bhh[idx];
}

// x [B,T,300] f32 -> xbf [B,T,320] bf16, zero-padded
__global__ void conv_x_kernel(const float* __restrict__ x, bf16* __restrict__ xbf) {
  size_t idx = (size_t)blockIdx.x * blockDim.x + threadIdx.x;
  const size_t total = (size_t)BATCH * SEQT * DINP1;
  if (idx >= total) return;
  int k = (int)(idx % DINP1);
  size_t bt = idx / DINP1;
  xbf[idx] = (k < DIN1) ? (bf16)x[bt * DIN1 + k] : (bf16)0.f;
}

__global__ void zero_u32_kernel(uint32* __restrict__ p, int n) {
  int idx = blockIdx.x * blockDim.x + threadIdx.x;
  if (idx < n) p[idx] = 0u;
}

// Load one 16x32 bf16 fragment: two b128 loads at dword offsets 0 and 8.
// Pair layout matches ISA K-map: element e -> K = ((e&8)<<1) + 8*(lane>=16) + (e&7),
// so consecutive (even,odd) elements are consecutive in memory.
__device__ __forceinline__ v16bf load_frag(const uint32* __restrict__ p) {
  const uint4 lo = *(const uint4*)(p);
  const uint4 hi = *(const uint4*)(p + 8);
  v8u u;
  u[0] = lo.x; u[1] = lo.y; u[2] = lo.z; u[3] = lo.w;
  u[4] = hi.x; u[5] = hi.y; u[6] = hi.z; u[7] = hi.w;
  return __builtin_bit_cast(v16bf, u);
}

// One fused LSTM timestep. Block = 128 threads = 4 wave32s = 4 gates of one
// (16 batch x 16 hidden) tile. Fully unrolled dual K-loop of bf16 WMMAs.
template <int DINPAD, int HPAD, int H, bool LAYER1>
__global__ __launch_bounds__(128)
void lstm_step_kernel(int t,
                      const bf16*  __restrict__ in_bf,     // [B,T,DINPAD]
                      const bf16*  __restrict__ WcatT,     // [4H, KROW]
                      const float* __restrict__ bias,      // [4H]
                      const bf16*  __restrict__ hp,        // [B,HPAD] (read)
                      bf16*        __restrict__ hn,        // [B,HPAD] (write)
                      float*       __restrict__ c_buf,     // [B,H]
                      const int*   __restrict__ lengths,   // LAYER1
                      bf16*        __restrict__ out1_bf16, // LAYER1
                      float*       __restrict__ hval_out,  // LAYER1
                      float*       __restrict__ out2)      // !LAYER1
{
  constexpr int N4H = 4 * H;
  __shared__ float Glds[4][16][16];

  const int tid  = threadIdx.x;
  const int wave = tid >> 5;            // gate: 0=i 1=f 2=g 3=o
  const int lane = tid & 31;
  const int m0 = blockIdx.x << 4;
  const int n0 = blockIdx.y << 4;
  const int mn  = lane & 15;
  const int hi8 = (lane >> 4) << 3;     // K-half select (A/B), M offset (C/D)

  // Clamp instead of branch: clamped rows/cols produce garbage that the
  // epilogue discards, keeping the K-loops branch-free (EXEC all-ones).
  int b = m0 + mn;               if (b > BATCH - 1) b = BATCH - 1;
  int col = wave * H + n0 + mn;  if (col > N4H - 1) col = N4H - 1;

  const uint32* __restrict__ Ain = (const uint32*)(in_bf + ((size_t)b * SEQT + t) * DINPAD) + (hi8 >> 1);
  const uint32* __restrict__ Ah  = (const uint32*)(hp + (size_t)b * HPAD) + (hi8 >> 1);
  const uint32* __restrict__ Bp  = (const uint32*)(WcatT + (size_t)col * KROW) + (hi8 >> 1);

  v8f acc = {};
#pragma unroll
  for (int k0 = 0; k0 < DINPAD; k0 += 32) {                  // input-segment GEMM
    v16bf a = load_frag(Ain + (k0 >> 1));
    v16bf w = load_frag(Bp + (k0 >> 1));
    acc = __builtin_amdgcn_wmma_f32_16x16x32_bf16(false, a, false, w, (short)0, acc, false, false);
  }
#pragma unroll
  for (int k0 = 0; k0 < HPAD; k0 += 32) {                    // recurrent-segment GEMM
    v16bf a = load_frag(Ah + (k0 >> 1));
    v16bf w = load_frag(Bp + ((DINPAD + k0) >> 1));
    acc = __builtin_amdgcn_wmma_f32_16x16x32_bf16(false, a, false, w, (short)0, acc, false, false);
  }

  // C/D layout: VGPR r -> M = r + 8*(lane>=16), N = lane&15.
#pragma unroll
  for (int r = 0; r < 8; ++r)
    Glds[wave][hi8 + r][mn] = acc[r];
  __syncthreads();

  // Fused cell update: 256 cells, 2 per thread.
  for (int idx = tid; idx < 256; idx += 128) {
    int mb = idx >> 4, nh = idx & 15;
    int bb = m0 + mb, h = n0 + nh;
    if (bb >= BATCH || h >= H) continue;
    float gi = Glds[0][mb][nh] + bias[0 * H + h];
    float gf = Glds[1][mb][nh] + bias[1 * H + h];
    float gg = Glds[2][mb][nh] + bias[2 * H + h];
    float go = Glds[3][mb][nh] + bias[3 * H + h];
    float i_ = sigmoidf_(gi);
    float f_ = sigmoidf_(gf);
    float g_ = tanhf(gg);
    float o_ = sigmoidf_(go);
    float c_old = c_buf[bb * H + h];
    float c_new = f_ * c_old + i_ * g_;
    float h_new = o_ * tanhf(c_new);
    c_buf[bb * H + h]  = c_new;
    hn[(size_t)bb * HPAD + h] = (bf16)h_new;
    if (LAYER1) {
      int len = lengths[bb];
      out1_bf16[((size_t)bb * SEQT + t) * H + h] = (t < len) ? (bf16)h_new : (bf16)0.f;
      if (t == len - 1) hval_out[bb * H + h] = h_new;   // last-valid-timestep gather
    } else {
      out2[((size_t)bb * SEQT + t) * H + h] = h_new;
    }
  }
}

extern "C" void kernel_launch(void* const* d_in, const int* in_sizes, int n_in,
                              void* d_out, int out_size, void* d_ws, size_t ws_size,
                              hipStream_t stream) {
  (void)in_sizes; (void)n_in; (void)out_size; (void)ws_size;
  const float* x       = (const float*)d_in[0];
  const int*   lengths = (const int*)  d_in[1];
  const float* Wih1    = (const float*)d_in[2];
  const float* Whh1    = (const float*)d_in[3];
  const float* bih1    = (const float*)d_in[4];
  const float* bhh1    = (const float*)d_in[5];
  const float* Wih2    = (const float*)d_in[6];
  const float* Whh2    = (const float*)d_in[7];
  const float* bih2    = (const float*)d_in[8];
  const float* bhh2    = (const float*)d_in[9];

  float* out2 = (float*)d_out;                               // [B,T,H2]
  float* hval = (float*)d_out + (size_t)BATCH * SEQT * HID2; // [B,H1]

  // ---- workspace layout (~119 MB, 256B-aligned bump allocator) ----
  char* ws = (char*)d_ws;
  size_t off = 0;
  auto take = [&](size_t bytes) -> char* {
    char* p = ws + off;
    off += (bytes + 255) & ~(size_t)255;
    return p;
  };
  bf16* WcatT1 = (bf16*)take((size_t)4 * HID1 * KROW * sizeof(bf16));
  bf16* WcatT2 = (bf16*)take((size_t)4 * HID2 * KROW * sizeof(bf16));
  float* bsum1 = (float*)take((size_t)4 * HID1 * sizeof(float));
  float* bsum2 = (float*)take((size_t)4 * HID2 * sizeof(float));
  bf16* xbf    = (bf16*)take((size_t)BATCH * SEQT * DINP1 * sizeof(bf16));
  bf16* out1   = (bf16*)take((size_t)BATCH * SEQT * HID1 * sizeof(bf16));
  // contiguous state block (zeroed each call): c1,c2 f32 then h ping-pongs bf16
  const size_t state_bytes = (size_t)BATCH * HID1 * 4 + (size_t)BATCH * HID2 * 4 +
                             2 * (size_t)BATCH * HPAD1 * 2 + 2 * (size_t)BATCH * HPAD2 * 2;
  char* sb = take(state_bytes);
  float* c1  = (float*)sb;
  float* c2  = c1 + BATCH * HID1;
  bf16*  h1a = (bf16*)(c2 + BATCH * HID2);
  bf16*  h1b = h1a + BATCH * HPAD1;
  bf16*  h2a = h1b + BATCH * HPAD1;
  bf16*  h2b = h2a + BATCH * HPAD2;

  // ---- setup ----
  {
    int tot1 = 4 * HID1 * KROW;
    prep_wcatT_kernel<<<(tot1 + 255) / 256, 256, 0, stream>>>(Wih1, Whh1, DIN1, DINP1, HID1, WcatT1);
    int tot2 = 4 * HID2 * KROW;
    prep_wcatT_kernel<<<(tot2 + 255) / 256, 256, 0, stream>>>(Wih2, Whh2, HID1, HID1, HID2, WcatT2);
    prep_bias_kernel<<<(4 * HID1 + 255) / 256, 256, 0, stream>>>(bih1, bhh1, 4 * HID1, bsum1);
    prep_bias_kernel<<<(4 * HID2 + 255) / 256, 256, 0, stream>>>(bih2, bhh2, 4 * HID2, bsum2);
    size_t xtot = (size_t)BATCH * SEQT * DINP1;
    conv_x_kernel<<<(unsigned)((xtot + 255) / 256), 256, 0, stream>>>(x, xbf);
    int zn = (int)(state_bytes / 4);
    zero_u32_kernel<<<(zn + 255) / 256, 256, 0, stream>>>((uint32*)sb, zn);
  }

  const dim3 blk(128);

  // ---- layer 1: 500 fused timesteps, bf16 h ping-pong ----
  {
    dim3 grid((BATCH + 15) / 16, (HID1 + 15) / 16);   // 13 x 16
    for (int t = 0; t < SEQT; ++t) {
      const bf16* hpv = (t & 1) ? h1b : h1a;
      bf16*       hnv = (t & 1) ? h1a : h1b;
      lstm_step_kernel<DINP1, HPAD1, HID1, true><<<grid, blk, 0, stream>>>(
          t, xbf, WcatT1, bsum1, hpv, hnv, c1, lengths, out1, hval, (float*)nullptr);
    }
  }

  // ---- layer 2: consumes masked bf16 out1, writes f32 out2 ----
  {
    dim3 grid((BATCH + 15) / 16, (HID2 + 15) / 16);   // 13 x 19
    for (int t = 0; t < SEQT; ++t) {
      const bf16* hpv = (t & 1) ? h2b : h2a;
      bf16*       hnv = (t & 1) ? h2a : h2b;
      lstm_step_kernel<HID1, HPAD2, HID2, false><<<grid, blk, 0, stream>>>(
          t, out1, WcatT2, bsum2, hpv, hnv, c2,
          (const int*)nullptr, (bf16*)nullptr, (float*)nullptr, out2);
    }
  }
}
